// LorentzMLR_71133248357031
// MI455X (gfx1250) — matrix operand order repaired and checked by reading, not compiled, not run on hardware
//
#include <hip/hip_runtime.h>
#include <hip/hip_bf16.h>
#include <cstdint>
#include <cstddef>

// Problem dims (from reference setup_inputs): B=1, S=2048, D=1024, C=50257
#define S_DIM 2048
#define D_DIM 1024
#define C_DIM 50257
#define EPS_F 1e-6f

typedef __attribute__((ext_vector_type(16))) __bf16 v16bf;
typedef __attribute__((ext_vector_type(8)))  float  v8f;

union Frag {
  v16bf bf;
  uint4 q[2];
};

__device__ __forceinline__ unsigned short f32_to_bf16_rne(float f) {
  unsigned u = __float_as_uint(f);
  u += 0x7FFFu + ((u >> 16) & 1u);
  return (unsigned short)(u >> 16);
}
__device__ __forceinline__ float bf16_bits_to_f32(unsigned short h) {
  return __uint_as_float(((unsigned)h) << 16);
}

// ---------------------------------------------------------------------------
// prep_w: per class row c, compute c0[c] = sqrt(1 + |W[c,1:]|^2) and split
// W[c,1:] into bf16 hi + bf16 lo residual. One wave (32 lanes) per row.
// ---------------------------------------------------------------------------
__global__ __launch_bounds__(256) void prep_w(const float* __restrict__ W,
                                              unsigned short* __restrict__ Whi,
                                              unsigned short* __restrict__ Wlo,
                                              float* __restrict__ c0) {
  const int wave = threadIdx.x >> 5;
  const int lane = threadIdx.x & 31;
  const int row  = blockIdx.x * 8 + wave;
  if (row >= C_DIM) return;
  const float* src = W + (size_t)row * (D_DIM + 1) + 1;
  unsigned short* hi = Whi + (size_t)row * D_DIM;
  unsigned short* lo = Wlo + (size_t)row * D_DIM;
  float ss = 0.0f;
  #pragma unroll 4
  for (int k = lane; k < D_DIM; k += 32) {
    float v = src[k];
    ss += v * v;
    unsigned short h = f32_to_bf16_rne(v);
    float r = v - bf16_bits_to_f32(h);
    hi[k] = h;
    lo[k] = f32_to_bf16_rne(r);
  }
  #pragma unroll
  for (int off = 16; off > 0; off >>= 1) ss += __shfl_xor(ss, off, 32);
  if (lane == 0) c0[row] = sqrtf(1.0f + ss);
}

// ---------------------------------------------------------------------------
// prep_x: same for x rows -> x0, xhi, xlo.
// ---------------------------------------------------------------------------
__global__ __launch_bounds__(256) void prep_x(const float* __restrict__ X,
                                              unsigned short* __restrict__ Xhi,
                                              unsigned short* __restrict__ Xlo,
                                              float* __restrict__ x0) {
  const int wave = threadIdx.x >> 5;
  const int lane = threadIdx.x & 31;
  const int row  = blockIdx.x * 8 + wave;
  if (row >= S_DIM) return;
  const float* src = X + (size_t)row * D_DIM;
  unsigned short* hi = Xhi + (size_t)row * D_DIM;
  unsigned short* lo = Xlo + (size_t)row * D_DIM;
  float ss = 0.0f;
  #pragma unroll 4
  for (int k = lane; k < D_DIM; k += 32) {
    float v = src[k];
    ss += v * v;
    unsigned short h = f32_to_bf16_rne(v);
    float r = v - bf16_bits_to_f32(h);
    hi[k] = h;
    lo[k] = f32_to_bf16_rne(r);
  }
  #pragma unroll
  for (int off = 16; off > 0; off >>= 1) ss += __shfl_xor(ss, off, 32);
  if (lane == 0) x0[row] = sqrtf(1.0f + ss);
}

// ---------------------------------------------------------------------------
// lorentz_gemm: out[s,c] = -acosh(max(x0[s]*c0[c] - dot(x[s],W[c]), 1+eps))
// Block = 256 threads = 8 waves in a 2(M) x 4(N) grid.
// Block tile: 128(M) x 128(N). Wave tile: 64(M) x 32(N) = 4x2 WMMA tiles.
// bf16x3 split product: acc += Ahi*Bhi + Alo*Bhi + Ahi*Blo  (f32 accum).
// Grid: (M blocks = 16) fastest, (N blocks = 393) outer -> W tile L2 reuse.
// ---------------------------------------------------------------------------
__global__ __launch_bounds__(256) void lorentz_gemm(
    const unsigned short* __restrict__ Whi, const unsigned short* __restrict__ Wlo,
    const float* __restrict__ c0,
    const unsigned short* __restrict__ Xhi, const unsigned short* __restrict__ Xlo,
    const float* __restrict__ x0,
    float* __restrict__ out) {
  const int lane  = threadIdx.x & 31;
  const int wave  = threadIdx.x >> 5;
  const int waveM = wave >> 2;   // 0..1
  const int waveN = wave & 3;    // 0..3
  const int M0 = blockIdx.x * 128 + waveM * 64;   // always < 2048
  const int N0 = blockIdx.y * 128 + waveN * 32;   // may run past C_DIM (ragged)

  const int lhalf = lane >> 4;   // 0 or 1
  const int l15   = lane & 15;

  // A rows (sequence positions) and B rows (classes, clamped for loads)
  const unsigned short* aHiRow[4];
  const unsigned short* aLoRow[4];
  #pragma unroll
  for (int mi = 0; mi < 4; ++mi) {
    const size_t r = (size_t)(M0 + mi * 16 + l15) * D_DIM;
    aHiRow[mi] = Xhi + r;
    aLoRow[mi] = Xlo + r;
  }
  int cls[2];
  const unsigned short* bHiRow[2];
  const unsigned short* bLoRow[2];
  #pragma unroll
  for (int ni = 0; ni < 2; ++ni) {
    int c = N0 + ni * 16 + l15;
    cls[ni] = c;
    int cc = c < C_DIM ? c : (C_DIM - 1);
    bHiRow[ni] = Whi + (size_t)cc * D_DIM;
    bLoRow[ni] = Wlo + (size_t)cc * D_DIM;
  }

  v8f acc[4][2];
  #pragma unroll
  for (int mi = 0; mi < 4; ++mi)
    #pragma unroll
    for (int ni = 0; ni < 2; ++ni)
      acc[mi][ni] = (v8f){0.f, 0.f, 0.f, 0.f, 0.f, 0.f, 0.f, 0.f};

  for (int k0 = 0; k0 < D_DIM; k0 += 32) {
    // A fragment (16x32 bf16): lane holds row M0+mi*16+l15,
    //   VGPR0..3 = K[kA..kA+7], VGPR4..7 = K[kA+16..kA+23], kA = k0 + 8*lhalf
    const int kA = k0 + (lhalf << 3);
    // B fragment (32x16 bf16): lane holds column class cls[ni],
    //   VGPR0..3 = K[kB..kB+7], VGPR4..7 = K[kB+16..kB+23], kB = k0 + 8*lhalf
    const int kB = k0 + (lhalf << 3);

    Frag ah[4], al[4], bh[2], bl[2];
    #pragma unroll
    for (int mi = 0; mi < 4; ++mi) {
      ah[mi].q[0] = *(const uint4*)(aHiRow[mi] + kA);
      ah[mi].q[1] = *(const uint4*)(aHiRow[mi] + kA + 16);
      al[mi].q[0] = *(const uint4*)(aLoRow[mi] + kA);
      al[mi].q[1] = *(const uint4*)(aLoRow[mi] + kA + 16);
    }
    #pragma unroll
    for (int ni = 0; ni < 2; ++ni) {
      bh[ni].q[0] = *(const uint4*)(bHiRow[ni] + kB);
      bh[ni].q[1] = *(const uint4*)(bHiRow[ni] + kB + 16);
      bl[ni].q[0] = *(const uint4*)(bLoRow[ni] + kB);
      bl[ni].q[1] = *(const uint4*)(bLoRow[ni] + kB + 16);
      // Prefetch next K-step of W (emits global_prefetch_b8)
      __builtin_prefetch(bHiRow[ni] + kB + 32, 0, 1);
      __builtin_prefetch(bLoRow[ni] + kB + 32, 0, 1);
    }

    #pragma unroll
    for (int mi = 0; mi < 4; ++mi) {
      #pragma unroll
      for (int ni = 0; ni < 2; ++ni) {
        v8f a = acc[mi][ni];
        a = __builtin_amdgcn_wmma_f32_16x16x32_bf16(false, ah[mi].bf, false, bh[ni].bf,
                                                    (short)0, a, false, false);
        a = __builtin_amdgcn_wmma_f32_16x16x32_bf16(false, al[mi].bf, false, bh[ni].bf,
                                                    (short)0, a, false, false);
        a = __builtin_amdgcn_wmma_f32_16x16x32_bf16(false, ah[mi].bf, false, bl[ni].bf,
                                                    (short)0, a, false, false);
        acc[mi][ni] = a;
      }
    }
  }

  // Epilogue: C/D layout -> element e of v8f is row M = e + 8*lhalf, col = l15.
  #pragma unroll
  for (int ni = 0; ni < 2; ++ni) {
    const int c = cls[ni];
    if (c >= C_DIM) continue;
    const float cc = c0[c];
    #pragma unroll
    for (int mi = 0; mi < 4; ++mi) {
      #pragma unroll
      for (int e = 0; e < 8; ++e) {
        const int s = M0 + mi * 16 + e + 8 * lhalf;
        float z = fmaf(x0[s], cc, -acc[mi][ni][e]);
        z = fmaxf(z, 1.0f + EPS_F);
        out[(size_t)s * C_DIM + c] = -acoshf(z);
      }
    }
  }
}

// ---------------------------------------------------------------------------
// Launch
// ---------------------------------------------------------------------------
extern "C" void kernel_launch(void* const* d_in, const int* in_sizes, int n_in,
                              void* d_out, int out_size, void* d_ws, size_t ws_size,
                              hipStream_t stream) {
  (void)in_sizes; (void)n_in; (void)out_size; (void)ws_size;
  const float* x = (const float*)d_in[0];   // (1, 2048, 1024) f32
  const float* W = (const float*)d_in[1];   // (50257, 1025)  f32
  float* out = (float*)d_out;               // (1, 2048, 50257) f32

  // Workspace layout (all 16B-aligned):
  //   Whi : C*D u16  = 102,926,336 B
  //   Wlo : C*D u16  = 102,926,336 B
  //   Xhi : S*D u16  =   4,194,304 B
  //   Xlo : S*D u16  =   4,194,304 B
  //   c0  : C   f32  =     201,028 B
  //   x0  : S   f32  =       8,192 B      (total ~214.3 MB)
  char* ws = (char*)d_ws;
  unsigned short* Whi = (unsigned short*)(ws);
  unsigned short* Wlo = (unsigned short*)(ws + (size_t)C_DIM * D_DIM * 2);
  unsigned short* Xhi = (unsigned short*)(ws + (size_t)C_DIM * D_DIM * 4);
  unsigned short* Xlo = (unsigned short*)(ws + (size_t)C_DIM * D_DIM * 4 + (size_t)S_DIM * D_DIM * 2);
  float* c0 = (float*)(ws + (size_t)C_DIM * D_DIM * 4 + (size_t)S_DIM * D_DIM * 4);
  float* x0 = (float*)((char*)c0 + (((size_t)C_DIM * 4 + 15) & ~(size_t)15));

  // 1) Split W -> bf16 hi/lo, compute c0.  8 rows per 256-thread block.
  prep_w<<<dim3((C_DIM + 7) / 8), dim3(256), 0, stream>>>(W, Whi, Wlo, c0);
  // 2) Split x -> bf16 hi/lo, compute x0.
  prep_x<<<dim3(S_DIM / 8), dim3(256), 0, stream>>>(x, Xhi, Xlo, x0);
  // 3) WMMA GEMM + acosh epilogue.
  //    grid.x = M blocks (16, fastest -> W-tile reuse in L2), grid.y = N blocks.
  lorentz_gemm<<<dim3(S_DIM / 128, (C_DIM + 127) / 128), dim3(256), 0, stream>>>(
      Whi, Wlo, c0, Xhi, Xlo, x0, out);
}
